// Interpolated_Conv_45715631898953
// MI455X (gfx1250) — matrix operand order repaired and checked by reading, not compiled, not run on hardware
//
#include <hip/hip_runtime.h>
#include <hip/hip_bf16.h>
#include <stdint.h>

#define SEQ      16384
#define IN_CH    32
#define OUT_CH   32
#define NK       11
#define KLEN     (2*SEQ + 1)          // 32769 taps
#define TAPS_PAD 32784                // 2049 * 16 (zero padded)
#define NCHUNK   2049
#define XROWS    (3*SEQ + 32)         // 49184 padded rows of x_pad
#define TT       256                  // t-tile per workgroup
#define PITCH    20                   // dwords per 32-half LDS row (TDM-padded, 16B aligned)

#define A_ROWS   (16*32)              // rows per A chunk (16 taps x 32 o)
#define X_ROWS   (TT + 16)            // rows per X window
#define ASZ_DW   (A_ROWS * PITCH)     // 10240 dwords
#define XSZ_DW   (X_ROWS * PITCH)     // 5440 dwords
#define LDS_BYTES ((2*ASZ_DW + 2*XSZ_DW) * 4)   // 125440 B

typedef __attribute__((ext_vector_type(16))) _Float16 v16h;
typedef __attribute__((ext_vector_type(8)))  float    v8f;
typedef __attribute__((ext_vector_type(4)))  unsigned int v4u;
typedef __attribute__((ext_vector_type(8)))  int      v8i;
typedef __attribute__((ext_vector_type(4)))  int      v4i;

union Frag { v16h v; uint32_t dw[8]; };

// ---------------------------------------------------------------------------
// TDM: async 2D tile load global -> LDS with hardware row padding.
// Tile: `rows` rows x 16 dwords; LDS gets 4 pad dwords appended per 16 dwords
// (pad_interval code 3 = 16 dw, pad_amount code 3 = 4 dw) -> 20-dword pitch.
// D# bitfields per CDNA5 ISA 8.3/8.4; data_size=2 (4-byte units); type=2.
// 6-arg builtin form (clang-23 / therock headers): extra zero group + cpol.
// ---------------------------------------------------------------------------
__device__ __forceinline__
void tdm_load_2d(uint32_t lds_byte_addr, const void* gptr, uint32_t rows)
{
  uint64_t ga = (uint64_t)(uintptr_t)gptr;
  v4u g0;
  g0.x = 1u;                                        // count=1, user mode
  g0.y = lds_byte_addr;                             // lds_addr (bytes)
  g0.z = (uint32_t)ga;                              // global_addr[31:0]
  g0.w = (uint32_t)(ga >> 32) | 0x80000000u;        // global_addr[56:32] | type=2
  v8i g1;
  g1[0] = (int)((2u << 16)    |                     // data_size = 4 bytes
                (1u << 20)    |                     // pad_enable
                (3u << 22)    |                     // pad_interval: 16 dwords
                (3u << 25));                        // pad_amount:   4 dwords
  g1[1] = (int)(16u << 16);                         // tensor_dim0 = 16 (bits 79:48)
  g1[2] = (int)((rows & 0xffffu) << 16);            // tensor_dim1 lo (bits 111:80)
  g1[3] = (int)(16u << 16);                         // tensor_dim1 hi=0 | tile_dim0=16
  g1[4] = (int)(rows & 0xffffu);                    // tile_dim1 = rows, tile_dim2=0 (2D)
  g1[5] = 16;                                       // tensor_dim0_stride = 16 units
  g1[6] = 0;                                        // dim0_stride hi | dim1_stride lo
  g1[7] = 0;                                        // dim1_stride hi (unused, 2D)
  v4i z4 = {0, 0, 0, 0};                            // groups 2/3 unused (2D tile)
  v8i z8 = {0, 0, 0, 0, 0, 0, 0, 0};                // extra group (unused)
  __builtin_amdgcn_tensor_load_to_lds(g0, g1, z4, z4, z8, 0);
}

// ---------------------------------------------------------------------------
// Long-kernel synthesis: un-normalized long kernel value at tap s (see ref).
// ---------------------------------------------------------------------------
__device__ __forceinline__
float longk(const float* __restrict__ kernels, const float* __restrict__ center,
            int o, int i, int s)
{
  if (s == SEQ) return center[o * IN_CH + i];
  int halfSide, p;
  if (s < SEQ) { halfSide = 0; p = SEQ - 1 - s; }     // flipped negative side
  else         { halfSide = 1; p = s - SEQ - 1; }     // positive side
  int q = (p >> 4) + 1;
  int l = 31 - __builtin_clz(q);                      // level index
  int start = 16 * ((1 << l) - 1);
  int u = p - start;
  float inv2l = 1.0f / (float)(1 << l);
  float coords = ((float)u + 0.5f) * inv2l - 0.5f;    // align_corners=False
  coords = fminf(fmaxf(coords, 0.0f), 15.0f);
  int i0 = (int)floorf(coords);
  if (i0 > 15) i0 = 15;
  int i1 = (i0 + 1 < 16) ? i0 + 1 : 15;
  float w = coords - (float)i0;
  const float* kr = kernels + ((size_t)(l * 2 * OUT_CH + halfSide * OUT_CH + o)
                               * IN_CH + i) * 16;
  float mult = exp2f(-(float)l / 4096.0f) * 0.9f + 0.1f;  // MULT[l]
  return mult * ((1.0f - w) * kr[i0] + w * kr[i1]);
}

// ---------------------------------------------------------------------------
// Stage 0: x [8][32][SEQ] f32  ->  xpad [8][XROWS][32] f16 (zero-padded)
// ---------------------------------------------------------------------------
__global__ __launch_bounds__(256)
void build_xpad(const float* __restrict__ x, _Float16* __restrict__ xpad)
{
  size_t idx = (size_t)blockIdx.x * 256 + threadIdx.x;   // 8*XROWS*32 exact
  int i = (int)(idx & 31);
  int t = (int)((idx >> 5) % XROWS);
  int b = (int)(idx / ((size_t)XROWS * 32));
  float v = 0.0f;
  int ts = t - SEQ;
  if (ts >= 0 && ts < SEQ)
    v = x[((size_t)b * IN_CH + i) * SEQ + ts];
  xpad[idx] = (_Float16)v;
}

// ---------------------------------------------------------------------------
// Stage 1: normalized long kernel -> Kmat [TAPS_PAD][OUT_CH][IN_CH] f16
// ---------------------------------------------------------------------------
__global__ __launch_bounds__(256)
void build_longk(const float* __restrict__ kernels,
                 const float* __restrict__ center,
                 _Float16* __restrict__ Kmat)
{
  const int o = blockIdx.x >> 5;
  const int i = blockIdx.x & 31;
  __shared__ float red[256];

  float ss = 0.0f;
  for (int s = threadIdx.x; s < KLEN; s += 256) {
    float v = longk(kernels, center, o, i, s);
    ss += v * v;
  }
  red[threadIdx.x] = ss;
  __syncthreads();
  for (int off = 128; off > 0; off >>= 1) {
    if (threadIdx.x < off) red[threadIdx.x] += red[threadIdx.x + off];
    __syncthreads();
  }
  const float inv = 1.0f / sqrtf(red[0]);

  for (int s = threadIdx.x; s < TAPS_PAD; s += 256) {
    float v = (s < KLEN) ? longk(kernels, center, o, i, s) * inv : 0.0f;
    Kmat[(size_t)s * (OUT_CH * IN_CH) + o * IN_CH + i] = (_Float16)v;
  }
}

// ---------------------------------------------------------------------------
// Stage 2: GEMM-conv, TDM double-buffered staging + v_wmma_f32_16x16x32_f16.
//   y[b,o,t] = sum_s sum_i Kmat[s][o][i] * xpad[b][t+s][i]
// Block: (b, 256-t tile). 8 waves = 2 o-blocks x 4 t-quadrants.
// Wave0 issues the A-tile TDM, wave1 the X-window TDM; everyone computes.
// LDS layout (dword offsets): A0 @ 0 | A1 @ ASZ_DW | X0 @ 2*ASZ_DW | X1 @ +XSZ_DW
// ---------------------------------------------------------------------------
__global__ __launch_bounds__(256)
void conv_wmma(const uint32_t* __restrict__ Kmat_dw,   // [TAPS_PAD][32][16 dw]
               const uint32_t* __restrict__ Xpad_dw,   // [8][XROWS][16 dw]
               float* __restrict__ out)
{
  extern __shared__ uint32_t smem[];

  const int tid   = threadIdx.x;
  const int lane  = tid & 31;
  const int wave  = tid >> 5;
  const int b     = blockIdx.x >> 6;
  const int t0    = (blockIdx.x & 63) * TT;
  const int ob    = wave & 1;                          // o block: rows ob*16..+15
  const int tq    = wave >> 1;                         // t quadrant: 64 t's
  const int hg    = lane >> 4;                         // half-group of wave
  const int l16   = lane & 15;

  const uint32_t* xb = Xpad_dw + (size_t)b * XROWS * 16;
  const uint32_t ldsBase = (uint32_t)(uintptr_t)&smem[0];   // LDS byte address

  v8f acc[4];
#pragma unroll
  for (int st = 0; st < 4; ++st) acc[st] = (v8f){0,0,0,0,0,0,0,0};

  // prologue: kick off chunk 0 into buffer 0
  if (wave == 0 && lane == 0)
    tdm_load_2d(ldsBase, Kmat_dw, A_ROWS);
  if (wave == 1 && lane == 0)
    tdm_load_2d(ldsBase + 2 * ASZ_DW * 4, xb + (size_t)t0 * 16, X_ROWS);

  for (int c = 0; c < NCHUNK; ++c) {
    const uint32_t cur  = (uint32_t)(c & 1);
    const uint32_t aOff = cur * ASZ_DW;                 // dwords
    const uint32_t xOff = 2u * ASZ_DW + cur * XSZ_DW;   // dwords

    if (wave < 2)
      __builtin_amdgcn_s_wait_tensorcnt((short)0);      // chunk c landed
    __syncthreads();                                    // visible to all waves;
                                                        // other buffer free again
    if (c + 1 < NCHUNK) {
      const int s1 = (c + 1) * 16;
      const uint32_t nxt = cur ^ 1u;
      if (wave == 0 && lane == 0)
        tdm_load_2d(ldsBase + nxt * ASZ_DW * 4,
                    Kmat_dw + (size_t)s1 * (32 * 16), A_ROWS);
      if (wave == 1 && lane == 0)
        tdm_load_2d(ldsBase + (2u * ASZ_DW + nxt * XSZ_DW) * 4,
                    xb + (size_t)(t0 + s1) * 16, X_ROWS);
    }

    const uint32_t* As = smem + aOff;
    const uint32_t* Xs = smem + xOff;
#pragma unroll 4
    for (int j = 0; j < 16; ++j) {
      // A fragment (16x32 f16, M=o_local, K=i):
      //  lane<16: M=lane, K pairs {2r,2r+1} (K=0..15); +16 for hg=1
      Frag a;
      const int rowA = (j * 32 + ob * 16 + l16) * PITCH;
#pragma unroll
      for (int r = 0; r < 8; ++r)
        a.dw[r] = As[rowA + 8 * hg + r];

#pragma unroll
      for (int st = 0; st < 4; ++st) {
        // B fragment (32x16 f16, K=i, N=t_local):
        //  lanes 0-15 hold K=0..15, lanes 16-31 hold K=16..31, col = lane&15
        Frag bf;
        const int rowX = (tq * 64 + st * 16 + l16 + j) * PITCH + 8 * hg;
#pragma unroll
        for (int r = 0; r < 8; ++r)
          bf.dw[r] = Xs[rowX + r];

        acc[st] = __builtin_amdgcn_wmma_f32_16x16x32_f16(
            false, a.v, false, bf.v, (short)0, acc[st], false, false);
      }
    }
    __syncthreads();                                    // done reading buf[cur]
  }

  // D layout: VGPR r -> M = r + 8*hg, N = lane&15
#pragma unroll
  for (int st = 0; st < 4; ++st) {
    const int t = t0 + tq * 64 + st * 16 + l16;
#pragma unroll
    for (int r = 0; r < 8; ++r) {
      const int o = ob * 16 + r + 8 * hg;
      out[((size_t)(b * OUT_CH + o)) * SEQ + t] = acc[st][r];
    }
  }
}

// ---------------------------------------------------------------------------
extern "C" void kernel_launch(void* const* d_in, const int* in_sizes, int n_in,
                              void* d_out, int out_size, void* d_ws, size_t ws_size,
                              hipStream_t stream)
{
  (void)in_sizes; (void)n_in; (void)out_size; (void)ws_size;
  const float* x       = (const float*)d_in[0];   // [8][32][SEQ]
  const float* kernels = (const float*)d_in[1];   // [11][64][32][16]
  const float* center  = (const float*)d_in[2];   // [32][32][1]
  float*       out     = (float*)d_out;           // [8][32][SEQ]

  _Float16* xpad = (_Float16*)d_ws;                        // 25.2 MB
  size_t xpad_elems = (size_t)8 * XROWS * IN_CH;           // 16B-aligned split
  _Float16* Kmat = xpad + xpad_elems;                      // 67.1 MB

  build_xpad<<<(8 * XROWS * IN_CH) / 256, 256, 0, stream>>>(x, xpad);
  build_longk<<<OUT_CH * IN_CH, 256, 0, stream>>>(kernels, center, Kmat);
  conv_wmma<<<8 * (SEQ / TT), 256, LDS_BYTES, stream>>>(
      (const uint32_t*)Kmat, (const uint32_t*)xpad, out);
}
